// GAT_77927886618858
// MI455X (gfx1250) — compile-verified, matrix operand order checked by reading, not run on hardware
//
#include <hip/hip_runtime.h>

typedef __attribute__((ext_vector_type(16))) __bf16 v16bf;
typedef __attribute__((ext_vector_type(8)))  float  v8f;

#define NNODES 100000
#define NEDGES 1600000
#define ETOT   (NNODES + NEDGES)

// ---------- helpers ----------
__device__ __forceinline__ __bf16 f2bf(float f) {
    return static_cast<__bf16>(f);   // native v_cvt (RNE) on gfx1250
}
// order-preserving float<->int encoding (involution) for atomicMax
__device__ __forceinline__ int f2ord(float f) {
    int i = __float_as_int(f);
    return (i >= 0) ? i : (i ^ 0x7fffffff);
}
__device__ __forceinline__ float ord2f(int i) {
    int j = (i >= 0) ? i : (i ^ 0x7fffffff);
    return __int_as_float(j);
}
__device__ __forceinline__ void edge_sd(const int* __restrict__ ei, int e, int& s, int& d) {
    if (e < NEDGES) { s = ei[e]; d = ei[NEDGES + e]; }
    else            { s = e - NEDGES; d = s; }       // appended self-loops
}

// ---------- W pre-swizzle into per-lane WMMA B-fragment order (f32 -> bf16) ----------
// fragment id = nt*4 + ks; storage: [frag][lane][16 bf16] -> one 32B load in the GEMM
__global__ void swizzle_w(const float* __restrict__ W, __bf16* __restrict__ out,
                          int ncols, int ntiles) {
    int gid = blockIdx.x * blockDim.x + threadIdx.x;
    int total = ntiles * 4 * 32;
    if (gid >= total) return;
    int lane = gid & 31, ks = (gid >> 5) & 3, nt = gid >> 7;
    int half = lane >> 4, koff = half * 8, col = nt * 16 + (lane & 15);
    __bf16* dst = out + (size_t)gid * 16;
#pragma unroll
    for (int v = 0; v < 8; ++v) {
        int k = 32 * ks + ((v < 4) ? (koff + 2 * v) : (16 + koff + 2 * (v - 4)));
        dst[2 * v]     = f2bf(W[(size_t)k * ncols + col]);
        dst[2 * v + 1] = f2bf(W[(size_t)(k + 1) * ncols + col]);
    }
}

// ---------- WMMA GEMM: Hout[M, NTILES*16] = X[M,128] @ W[128, NTILES*16] ----------
// 8 waves / block, each wave owns a 16-row strip; A fragments held in registers and
// reused across all N-tiles; NTILES is compile-time so all accumulator chains are
// independent and can be interleaved by the scheduler (hides WMMA->WMMA hazards).
template <int NTILES>
__global__ void gemm_bf16_wmma(const float* __restrict__ X,
                               const __bf16* __restrict__ Wsw,
                               float* __restrict__ Hout,
                               int M) {
    constexpr int ncols = NTILES * 16;
    const int lane  = threadIdx.x & 31;
    const int wave  = threadIdx.x >> 5;
    const int row0  = (blockIdx.x * 8 + wave) * 16;
    if (row0 >= M) return;                           // wave-uniform exit
    const int half  = lane >> 4;
    const int koff  = half * 8;
    const int mlane = lane & 15;
    int rowA = row0 + mlane;
    if (rowA >= M) rowA = M - 1;                     // clamp loads; stores handled below
    const float* xrow = X + (size_t)rowA * 128;

    v16bf afrag[4];
#pragma unroll
    for (int ks = 0; ks < 4; ++ks) {
#pragma unroll
        for (int v = 0; v < 8; ++v) {
            int k = 32 * ks + ((v < 4) ? (koff + 2 * v) : (16 + koff + 2 * (v - 4)));
            float2 p = *(const float2*)(xrow + k);
            afrag[ks][2 * v]     = f2bf(p.x);
            afrag[ks][2 * v + 1] = f2bf(p.y);
        }
    }

    v8f acc[NTILES];
#pragma unroll
    for (int nt = 0; nt < NTILES; ++nt) acc[nt] = {};

#pragma unroll
    for (int nt = 0; nt < NTILES; ++nt) {
#pragma unroll
        for (int ks = 0; ks < 4; ++ks) {
            v16bf b = *(const v16bf*)(Wsw + ((size_t)((nt * 4 + ks) * 32 + lane)) * 16);
            acc[nt] = __builtin_amdgcn_wmma_f32_16x16x32_bf16(
                false, afrag[ks], false, b, (short)0, acc[nt], false, false);
        }
    }

    // C layout: vgpr r -> (M = row0 + r + half*8, N = nt*16 + mlane)
    float* hbase = Hout + (size_t)(row0 + half * 8) * ncols + mlane;
    if (row0 + 16 <= M) {                            // wave-uniform fast path
#pragma unroll
        for (int nt = 0; nt < NTILES; ++nt)
#pragma unroll
            for (int r = 0; r < 8; ++r)
                hbase[(size_t)r * ncols + nt * 16] = acc[nt][r];
    } else {                                          // partial tail block
#pragma unroll
        for (int nt = 0; nt < NTILES; ++nt)
#pragma unroll
            for (int r = 0; r < 8; ++r) {
                int mrow = row0 + r + half * 8;
                if (mrow < M) Hout[(size_t)mrow * ncols + nt * 16 + mlane] = acc[nt][r];
            }
    }
}

// ---------- per-node attention logits: as[n,h] = h[n,h,:]·att_src[h], ad likewise ----------
__global__ void node_alpha(const float* __restrict__ h, const float* __restrict__ att_s,
                           const float* __restrict__ att_d, float* __restrict__ as,
                           float* __restrict__ ad, int hshift, int chead) {
    int gid = blockIdx.x * blockDim.x + threadIdx.x;
    int H = 1 << hshift;
    if (gid >= NNODES * H) return;
    int n = gid >> hshift, hh = gid & (H - 1);
    int C = H * chead;
    const float* hp = h + (size_t)n * C + hh * chead;
    const float* sp = att_s + hh * chead;
    const float* dp = att_d + hh * chead;
    float s0 = 0.f, s1 = 0.f;
    for (int c = 0; c < chead; ++c) { float v = hp[c]; s0 += v * sp[c]; s1 += v * dp[c]; }
    as[gid] = s0; ad[gid] = s1;
}

__global__ void fill_u32(unsigned* __restrict__ p, unsigned v, int n) {
    int i = blockIdx.x * blockDim.x + threadIdx.x;
    if (i < n) p[i] = v;
}

// ---------- edge pass 1: leaky_relu logit + segment max (ordered-int atomicMax) ----------
__global__ void edge_alpha_max(const int* __restrict__ ei, const float* __restrict__ as,
                               const float* __restrict__ ad, float* __restrict__ albuf,
                               int* __restrict__ amax, int H) {
    int e = blockIdx.x * blockDim.x + threadIdx.x;
    if (e >= ETOT) return;
    int s, d; edge_sd(ei, e, s, d);
    for (int hh = 0; hh < H; ++hh) {
        float a = as[(size_t)s * H + hh] + ad[(size_t)d * H + hh];
        a = (a > 0.f) ? a : 0.2f * a;                // leaky_relu, slope 0.2
        albuf[(size_t)e * H + hh] = a;
        atomicMax(&amax[(size_t)d * H + hh], f2ord(a));
    }
}

// ---------- edge pass 2: exp(alpha - max) + segment sum ----------
__global__ void edge_exp_sum(const int* __restrict__ ei, float* __restrict__ albuf,
                             const int* __restrict__ amax, float* __restrict__ den, int H) {
    int e = blockIdx.x * blockDim.x + threadIdx.x;
    if (e >= ETOT) return;
    int s, d; edge_sd(ei, e, s, d);
    for (int hh = 0; hh < H; ++hh) {
        float m  = ord2f(amax[(size_t)d * H + hh]);
        float ex = __expf(albuf[(size_t)e * H + hh] - m);
        albuf[(size_t)e * H + hh] = ex;
        atomicAdd(&den[(size_t)d * H + hh], ex);
    }
}

// ---------- edge pass 3: out[dst] += h[src] * (ex/denom)  (L2-resident f32 atomics) ----------
__global__ void edge_aggregate(const int* __restrict__ ei, const float* __restrict__ exbuf,
                               const float* __restrict__ den, const float* __restrict__ hsrc,
                               float* __restrict__ out, int cshift, int headshift, int H) {
    long gid = (long)blockIdx.x * blockDim.x + threadIdx.x;
    long total = (long)ETOT << cshift;
    if (gid >= total) return;
    int C  = 1 << cshift;
    int e  = (int)(gid >> cshift);
    int c  = (int)gid & (C - 1);
    int hh = c >> headshift;
    int s, d; edge_sd(ei, e, s, d);
    float ex = exbuf[(size_t)e * H + hh];
    float dn = den[(size_t)d * H + hh];
    float w  = ex / (dn + 1e-16f);
    atomicAdd(&out[(size_t)d * C + c], hsrc[(size_t)s * C + c] * w);
}

// ---------- ELU(x + b1) ----------
__global__ void elu_bias(const float* __restrict__ in, const float* __restrict__ b,
                         float* __restrict__ out) {
    int i = blockIdx.x * blockDim.x + threadIdx.x;
    if (i >= NNODES * 128) return;
    float v = in[i] + b[i & 127];
    out[i] = (v > 0.f) ? v : (__expf(v) - 1.f);
}

// ---------- out += b2 ----------
__global__ void bias_add(float* __restrict__ out, const float* __restrict__ b) {
    int i = blockIdx.x * blockDim.x + threadIdx.x;
    if (i >= NNODES * 16) return;
    out[i] += b[i & 15];
}

extern "C" void kernel_launch(void* const* d_in, const int* in_sizes, int n_in,
                              void* d_out, int out_size, void* d_ws, size_t ws_size,
                              hipStream_t stream) {
    (void)in_sizes; (void)n_in; (void)out_size; (void)ws_size;
    const float* x   = (const float*)d_in[0];
    const int*   ei  = (const int*)d_in[1];     // [2, E] int32
    const float* W1  = (const float*)d_in[2];
    const float* b1  = (const float*)d_in[3];
    const float* as1 = (const float*)d_in[4];
    const float* ad1 = (const float*)d_in[5];
    const float* W2  = (const float*)d_in[6];
    const float* b2  = (const float*)d_in[7];
    const float* as2 = (const float*)d_in[8];
    const float* ad2 = (const float*)d_in[9];
    float* out = (float*)d_out;

    char* ws = (char*)d_ws;
    size_t off = 0;
    auto alloc = [&](size_t bytes) -> char* {
        char* p = ws + off;
        off = (off + bytes + 255) & ~((size_t)255);
        return p;
    };
    float*  h1    = (float*)alloc((size_t)NNODES * 128 * 4);
    float*  out1  = (float*)alloc((size_t)NNODES * 128 * 4);
    float*  asrc1 = (float*)alloc((size_t)NNODES * 4 * 4);
    float*  adst1 = (float*)alloc((size_t)NNODES * 4 * 4);
    int*    amax1 = (int*)  alloc((size_t)NNODES * 4 * 4);
    float*  den1  = (float*)alloc((size_t)NNODES * 4 * 4);
    float*  alb1  = (float*)alloc((size_t)ETOT * 4 * 4);
    float*  h2    = (float*)alloc((size_t)NNODES * 16 * 4);
    float*  asrc2 = (float*)alloc((size_t)NNODES * 4);
    float*  adst2 = (float*)alloc((size_t)NNODES * 4);
    int*    amax2 = (int*)  alloc((size_t)NNODES * 4);
    float*  den2  = (float*)alloc((size_t)NNODES * 4);
    float*  alb2  = (float*)alloc((size_t)ETOT * 4);
    __bf16* w1sw  = (__bf16*)alloc((size_t)128 * 128 * 2);
    __bf16* w2sw  = (__bf16*)alloc((size_t)128 * 16 * 2);

    const int B = 256;
    auto g = [](long n, int b) { return (unsigned)((n + b - 1) / b); };

    // weights -> bf16 WMMA fragment order
    swizzle_w<<<g(8 * 4 * 32, B), B, 0, stream>>>(W1, w1sw, 128, 8);
    swizzle_w<<<1, 128, 0, stream>>>(W2, w2sw, 16, 1);

    // ---- layer 1 ----
    gemm_bf16_wmma<8><<<g(NNODES, 128), 256, 0, stream>>>(x, w1sw, h1, NNODES);
    node_alpha<<<g((long)NNODES * 4, B), B, 0, stream>>>(h1, as1, ad1, asrc1, adst1, 2, 32);
    fill_u32<<<g((long)NNODES * 4, B), B, 0, stream>>>((unsigned*)amax1, 0x80000000u, NNODES * 4);
    fill_u32<<<g((long)NNODES * 4, B), B, 0, stream>>>((unsigned*)den1, 0u, NNODES * 4);
    fill_u32<<<g((long)NNODES * 128, B), B, 0, stream>>>((unsigned*)out1, 0u, NNODES * 128);
    edge_alpha_max<<<g(ETOT, B), B, 0, stream>>>(ei, asrc1, adst1, alb1, amax1, 4);
    edge_exp_sum  <<<g(ETOT, B), B, 0, stream>>>(ei, alb1, amax1, den1, 4);
    edge_aggregate<<<g((long)ETOT * 128, B), B, 0, stream>>>(ei, alb1, den1, h1, out1, 7, 5, 4);
    elu_bias<<<g((long)NNODES * 128, B), B, 0, stream>>>(out1, b1, h1);   // h1 := elu(out1+b1)

    // ---- layer 2 ----
    gemm_bf16_wmma<1><<<g(NNODES, 128), 256, 0, stream>>>(h1, w2sw, h2, NNODES);
    node_alpha<<<g(NNODES, B), B, 0, stream>>>(h2, as2, ad2, asrc2, adst2, 0, 16);
    fill_u32<<<g(NNODES, B), B, 0, stream>>>((unsigned*)amax2, 0x80000000u, NNODES);
    fill_u32<<<g(NNODES, B), B, 0, stream>>>((unsigned*)den2, 0u, NNODES);
    fill_u32<<<g((long)NNODES * 16, B), B, 0, stream>>>((unsigned*)out, 0u, NNODES * 16);
    edge_alpha_max<<<g(ETOT, B), B, 0, stream>>>(ei, asrc2, adst2, alb2, amax2, 1);
    edge_exp_sum  <<<g(ETOT, B), B, 0, stream>>>(ei, alb2, amax2, den2, 1);
    edge_aggregate<<<g((long)ETOT * 16, B), B, 0, stream>>>(ei, alb2, den2, h2, out, 4, 4, 1);
    bias_add<<<g((long)NNODES * 16, B), B, 0, stream>>>(out, b2);
}
